// InputFeedRNNDecoderLayers_11785390260428
// MI455X (gfx1250) — compile-verified, matrix operand order checked by reading, not run on hardware
//
#include <hip/hip_runtime.h>
#include <hip/hip_bf16.h>

typedef __bf16 bf16_t;
typedef __attribute__((ext_vector_type(16))) __bf16 v16bf;
typedef __attribute__((ext_vector_type(8)))  __bf16 v8bf;
typedef __attribute__((ext_vector_type(8)))  float  v8f;

#define T_LEN 64
#define B_SZ  128
#define S_LEN 64
#define E_DIM 512
#define H_DIM 1024
#define G_DIM 4096   // 4*H
#define EH_DIM 1536  // E+H
#define H2_DIM 2048  // 2*H

// ---------------------------------------------------------------------------
// bf16 WMMA GEMM:  C[M=128, N] (+)= A[128, K](bf16, lda) @ W[N, K](bf16, ldw)^T
// grid.x = N/64, block = 256 (8 waves). Wave w covers rows m0 = 16*w (full M),
// and a 64-wide N strip; 4 accumulators reuse one A fragment per k-step.
// Per ISA 7.12.2 (16-bit A 16x32): lane l (m = l&15, half = l>>4) holds
// K = half*8 .. +7 and K = 16+half*8 .. +7  -> two contiguous 16B loads.
// B operand (K x N) with lane = n needs column n of B = row n of W: contiguous.
// ---------------------------------------------------------------------------
__global__ __launch_bounds__(256)
void gemm_bf16_wmma(const bf16_t* __restrict__ A, int lda,
                    const bf16_t* __restrict__ W, int ldw,
                    float* __restrict__ C, int ldc,
                    int K, int accumulate)
{
    const int lane = threadIdx.x & 31;
    const int wave = threadIdx.x >> 5;
    const int m0   = wave << 4;              // 8 waves * 16 = M = 128
    const int n0   = blockIdx.x << 6;        // 64-wide N strip
    const int r    = lane & 15;
    const int half = lane >> 4;

    const bf16_t* arow = A + (long)(m0 + r) * lda;

    v8f acc[4];
    if (accumulate) {
#pragma unroll
        for (int j = 0; j < 4; ++j) {
            const int n = n0 + j * 16 + r;
#pragma unroll
            for (int rr = 0; rr < 8; ++rr)
                acc[j][rr] = C[(long)(m0 + half * 8 + rr) * ldc + n];
        }
    } else {
        v8f z = {};
#pragma unroll
        for (int j = 0; j < 4; ++j) acc[j] = z;
    }

    for (int k0 = 0; k0 < K; k0 += 32) {
        union { v16bf v; v8bf h[2]; } a;
        a.h[0] = *(const v8bf*)(arow + k0 + half * 8);
        a.h[1] = *(const v8bf*)(arow + k0 + 16 + half * 8);
#pragma unroll
        for (int j = 0; j < 4; ++j) {
            const bf16_t* wrow = W + (long)(n0 + j * 16 + r) * ldw;
            union { v16bf v; v8bf h[2]; } b;
            b.h[0] = *(const v8bf*)(wrow + k0 + half * 8);
            b.h[1] = *(const v8bf*)(wrow + k0 + 16 + half * 8);
            acc[j] = __builtin_amdgcn_wmma_f32_16x16x32_bf16(
                false, a.v, false, b.v, (short)0, acc[j], false, false);
        }
    }

    // C/D layout: lane (l&15) = n, VGPR rr -> m = 8*half + rr
#pragma unroll
    for (int j = 0; j < 4; ++j) {
        const int n = n0 + j * 16 + r;
#pragma unroll
        for (int rr = 0; rr < 8; ++rr)
            C[(long)(m0 + half * 8 + rr) * ldc + n] = acc[j][rr];
    }
}

// ---------------------------------------------------------------------------
// Elementwise / setup kernels
// ---------------------------------------------------------------------------
__global__ void k_f32_to_bf16(const float* __restrict__ in, bf16_t* __restrict__ out, int n)
{
    int i = blockIdx.x * blockDim.x + threadIdx.x;
    if (i < n) out[i] = (bf16_t)in[i];
}

__global__ void k_copy_f32(const float* __restrict__ in, float* __restrict__ out, int n)
{
    int i = blockIdx.x * blockDim.x + threadIdx.x;
    if (i < n) out[i] = in[i];
}

__device__ __forceinline__ float sigf(float x) { return 1.0f / (1.0f + __expf(-x)); }

// gates [B, 4H] (i,f,g,o), biases [4H]; updates c (fp32, in place), writes h (bf16)
__global__ void k_lstm_act(const float* __restrict__ gates,
                           const float* __restrict__ bih,
                           const float* __restrict__ bhh,
                           float* __restrict__ c, bf16_t* __restrict__ hbf)
{
    int idx = blockIdx.x * blockDim.x + threadIdx.x;      // B*H threads
    if (idx >= B_SZ * H_DIM) return;
    int b = idx >> 10;
    int h = idx & (H_DIM - 1);
    const float* g = gates + (long)b * G_DIM;
    float gi = g[h]              + bih[h]              + bhh[h];
    float gf = g[H_DIM + h]      + bih[H_DIM + h]      + bhh[H_DIM + h];
    float gg = g[2 * H_DIM + h]  + bih[2 * H_DIM + h]  + bhh[2 * H_DIM + h];
    float go = g[3 * H_DIM + h]  + bih[3 * H_DIM + h]  + bhh[3 * H_DIM + h];
    float cn = sigf(gf) * c[idx] + sigf(gi) * tanhf(gg);
    float hn = sigf(go) * tanhf(cn);
    c[idx]   = cn;
    hbf[idx] = (bf16_t)hn;
}

// scores[b,s] = q[b,:] . ctx[s,b,:]   (ctx is [S,B,H] fp32)
__global__ void k_scores(const float* __restrict__ q, const float* __restrict__ ctx,
                         float* __restrict__ scores)
{
    int idx = blockIdx.x * blockDim.x + threadIdx.x;      // B*S threads
    if (idx >= B_SZ * S_LEN) return;
    int b = idx >> 6;
    int s = idx & (S_LEN - 1);
    const float4* qp = (const float4*)(q + (long)b * H_DIM);
    const float4* cp = (const float4*)(ctx + ((long)s * B_SZ + b) * H_DIM);
    float acc = 0.f;
#pragma unroll 4
    for (int i = 0; i < H_DIM / 4; ++i) {
        float4 a = qp[i], c2 = cp[i];
        acc += a.x * c2.x + a.y * c2.y + a.z * c2.z + a.w * c2.w;
    }
    scores[idx] = acc;
}

// One wave32 per batch row; S=64 -> 2 elements per lane. Writes align (ws) and attns out.
__global__ void k_softmax(const float* __restrict__ scores,
                          float* __restrict__ align, float* __restrict__ attn_out)
{
    int b = blockIdx.x;
    int lane = threadIdx.x;                                // 0..31
    float v0 = scores[b * S_LEN + lane];
    float v1 = scores[b * S_LEN + 32 + lane];
    float m = fmaxf(v0, v1);
#pragma unroll
    for (int off = 16; off > 0; off >>= 1) m = fmaxf(m, __shfl_xor(m, off));
    float e0 = __expf(v0 - m), e1 = __expf(v1 - m);
    float s = e0 + e1;
#pragma unroll
    for (int off = 16; off > 0; off >>= 1) s += __shfl_xor(s, off);
    float inv = 1.0f / s;
    float a0 = e0 * inv, a1 = e1 * inv;
    align[b * S_LEN + lane]       = a0;
    align[b * S_LEN + 32 + lane]  = a1;
    attn_out[b * S_LEN + lane]      = a0;
    attn_out[b * S_LEN + 32 + lane] = a1;
}

// cvec[b,h] = sum_s align[b,s]*ctx[s,b,h]; builds concat[b] = [cvec | h1] in bf16
__global__ void k_cvec_concat(const float* __restrict__ align, const float* __restrict__ ctx,
                              const bf16_t* __restrict__ h1bf, bf16_t* __restrict__ cat)
{
    int idx = blockIdx.x * blockDim.x + threadIdx.x;      // B*H threads
    if (idx >= B_SZ * H_DIM) return;
    int b = idx >> 10;
    int h = idx & (H_DIM - 1);
    float acc = 0.f;
#pragma unroll 4
    for (int s = 0; s < S_LEN; ++s)
        acc += align[b * S_LEN + s] * ctx[((long)s * B_SZ + b) * H_DIM + h];
    cat[(long)b * H2_DIM + h]         = (bf16_t)acc;
    cat[(long)b * H2_DIM + H_DIM + h] = h1bf[idx];
}

// attn_h = tanh(ao); write fp32 output slice and bf16 input-feed for next step
__global__ void k_tanh_out(const float* __restrict__ ao, float* __restrict__ out_t,
                           bf16_t* __restrict__ feedbf)
{
    int idx = blockIdx.x * blockDim.x + threadIdx.x;      // B*H threads
    if (idx >= B_SZ * H_DIM) return;
    float v = tanhf(ao[idx]);
    out_t[idx]  = v;
    feedbf[idx] = (bf16_t)v;
}

// ---------------------------------------------------------------------------
extern "C" void kernel_launch(void* const* d_in, const int* in_sizes, int n_in,
                              void* d_out, int out_size, void* d_ws, size_t ws_size,
                              hipStream_t stream)
{
    (void)in_sizes; (void)n_in; (void)out_size; (void)ws_size;

    const float* emb      = (const float*)d_in[0];   // [T,B,E]
    const float* ctx      = (const float*)d_in[1];   // [S,B,H]
    const float* ifeed    = (const float*)d_in[2];   // [B,H]
    const float* h0_in    = (const float*)d_in[3];   // [2,B,H]
    const float* c0_in    = (const float*)d_in[4];   // [2,B,H]
    const float* W_ih0    = (const float*)d_in[5];   // [4H, E+H]
    const float* W_hh0    = (const float*)d_in[6];   // [4H, H]
    const float* b_ih0    = (const float*)d_in[7];
    const float* b_hh0    = (const float*)d_in[8];
    const float* W_ih1    = (const float*)d_in[9];   // [4H, H]
    const float* W_hh1    = (const float*)d_in[10];
    const float* b_ih1    = (const float*)d_in[11];
    const float* b_hh1    = (const float*)d_in[12];
    const float* W_ain    = (const float*)d_in[13];  // [H, H]
    const float* W_aout   = (const float*)d_in[14];  // [H, 2H]

    float* out = (float*)d_out;                       // [T,B,H] then [T,B,S]
    const size_t OUT_H_TOTAL = (size_t)T_LEN * B_SZ * H_DIM;

    // ---- workspace carve-out (all sizes 256B aligned) ----
    char*  ws  = (char*)d_ws;
    size_t off = 0;
    auto alloc = [&](size_t bytes) -> void* {
        void* p = ws + off;
        off = (off + bytes + 255) & ~(size_t)255;
        return p;
    };
    bf16_t* wih0bf = (bf16_t*)alloc((size_t)G_DIM * EH_DIM * 2);
    bf16_t* whh0bf = (bf16_t*)alloc((size_t)G_DIM * H_DIM * 2);
    bf16_t* wih1bf = (bf16_t*)alloc((size_t)G_DIM * H_DIM * 2);
    bf16_t* whh1bf = (bf16_t*)alloc((size_t)G_DIM * H_DIM * 2);
    bf16_t* wainbf = (bf16_t*)alloc((size_t)H_DIM * H_DIM * 2);
    bf16_t* waoutbf= (bf16_t*)alloc((size_t)H_DIM * H2_DIM * 2);
    bf16_t* embbf  = (bf16_t*)alloc((size_t)T_LEN * B_SZ * E_DIM * 2);
    bf16_t* feedbf = (bf16_t*)alloc((size_t)B_SZ * H_DIM * 2);
    bf16_t* h0bf   = (bf16_t*)alloc((size_t)B_SZ * H_DIM * 2);
    bf16_t* h1bf   = (bf16_t*)alloc((size_t)B_SZ * H_DIM * 2);
    bf16_t* catbf  = (bf16_t*)alloc((size_t)B_SZ * H2_DIM * 2);
    float*  c0f    = (float*) alloc((size_t)B_SZ * H_DIM * 4);
    float*  c1f    = (float*) alloc((size_t)B_SZ * H_DIM * 4);
    float*  gates  = (float*) alloc((size_t)B_SZ * G_DIM * 4);
    float*  qbuf   = (float*) alloc((size_t)B_SZ * H_DIM * 4);
    float*  aobuf  = (float*) alloc((size_t)B_SZ * H_DIM * 4);
    float*  scores = (float*) alloc((size_t)B_SZ * S_LEN * 4);
    float*  align  = (float*) alloc((size_t)B_SZ * S_LEN * 4);

    const int TB = 256;
    auto cvt = [&](const float* src, bf16_t* dst, size_t n) {
        k_f32_to_bf16<<<(unsigned)((n + TB - 1) / TB), TB, 0, stream>>>(src, dst, (int)n);
    };

    // ---- one-time (per call) conversions: weights, embeddings, state init ----
    cvt(W_ih0,  wih0bf,  (size_t)G_DIM * EH_DIM);
    cvt(W_hh0,  whh0bf,  (size_t)G_DIM * H_DIM);
    cvt(W_ih1,  wih1bf,  (size_t)G_DIM * H_DIM);
    cvt(W_hh1,  whh1bf,  (size_t)G_DIM * H_DIM);
    cvt(W_ain,  wainbf,  (size_t)H_DIM * H_DIM);
    cvt(W_aout, waoutbf, (size_t)H_DIM * H2_DIM);
    cvt(emb,    embbf,   (size_t)T_LEN * B_SZ * E_DIM);
    cvt(ifeed,  feedbf,  (size_t)B_SZ * H_DIM);
    cvt(h0_in,                 h0bf, (size_t)B_SZ * H_DIM);
    cvt(h0_in + B_SZ * H_DIM,  h1bf, (size_t)B_SZ * H_DIM);
    k_copy_f32<<<(B_SZ * H_DIM) / TB, TB, 0, stream>>>(c0_in, c0f, B_SZ * H_DIM);
    k_copy_f32<<<(B_SZ * H_DIM) / TB, TB, 0, stream>>>(c0_in + B_SZ * H_DIM, c1f, B_SZ * H_DIM);

    const int BH_BLOCKS = (B_SZ * H_DIM) / TB;           // 512
    const int BS_BLOCKS = (B_SZ * S_LEN) / TB;           // 32

    for (int t = 0; t < T_LEN; ++t) {
        const bf16_t* embt = embbf + (size_t)t * B_SZ * E_DIM;

        // LSTM layer 0: gates = emb_t@Wih0[:, :E]^T + feed@Wih0[:, E:]^T + h0@Whh0^T
        gemm_bf16_wmma<<<G_DIM / 64, 256, 0, stream>>>(embt,  E_DIM, wih0bf,         EH_DIM, gates, G_DIM, E_DIM, 0);
        gemm_bf16_wmma<<<G_DIM / 64, 256, 0, stream>>>(feedbf, H_DIM, wih0bf + E_DIM, EH_DIM, gates, G_DIM, H_DIM, 1);
        gemm_bf16_wmma<<<G_DIM / 64, 256, 0, stream>>>(h0bf,  H_DIM, whh0bf,         H_DIM,  gates, G_DIM, H_DIM, 1);
        k_lstm_act<<<BH_BLOCKS, TB, 0, stream>>>(gates, b_ih0, b_hh0, c0f, h0bf);

        // LSTM layer 1
        gemm_bf16_wmma<<<G_DIM / 64, 256, 0, stream>>>(h0bf, H_DIM, wih1bf, H_DIM, gates, G_DIM, H_DIM, 0);
        gemm_bf16_wmma<<<G_DIM / 64, 256, 0, stream>>>(h1bf, H_DIM, whh1bf, H_DIM, gates, G_DIM, H_DIM, 1);
        k_lstm_act<<<BH_BLOCKS, TB, 0, stream>>>(gates, b_ih1, b_hh1, c1f, h1bf);

        // Attention (general): q = h1 @ W_attn_in^T
        gemm_bf16_wmma<<<H_DIM / 64, 256, 0, stream>>>(h1bf, H_DIM, wainbf, H_DIM, qbuf, H_DIM, H_DIM, 0);
        k_scores<<<BS_BLOCKS, TB, 0, stream>>>(qbuf, ctx, scores);
        k_softmax<<<B_SZ, 32, 0, stream>>>(scores, align,
                                           out + OUT_H_TOTAL + (size_t)t * B_SZ * S_LEN);
        k_cvec_concat<<<BH_BLOCKS, TB, 0, stream>>>(align, ctx, h1bf, catbf);

        // attn_h = tanh([cvec | h1] @ W_attn_out^T); also becomes next input-feed
        gemm_bf16_wmma<<<H_DIM / 64, 256, 0, stream>>>(catbf, H2_DIM, waoutbf, H2_DIM, aobuf, H_DIM, H2_DIM, 0);
        k_tanh_out<<<BH_BLOCKS, TB, 0, stream>>>(aobuf, out + (size_t)t * B_SZ * H_DIM, feedbf);
    }
}
